// Variacne_Block_46875273069273
// MI455X (gfx1250) — matrix operand order repaired and checked by reading, not compiled, not run on hardware
//
#include <hip/hip_runtime.h>
#include <hip/hip_bf16.h>

typedef __bf16 bf16_t;
typedef __attribute__((ext_vector_type(16))) __bf16 v16bf;
typedef __attribute__((ext_vector_type(8)))  __bf16 v8bf;
typedef __attribute__((ext_vector_type(8)))  float  v8f;

#define BB   8
#define CC   512
#define NH   8
#define HD   64
#define TPP  512
#define TTT  256
#define TUU  2048
#define STK  4

__device__ __forceinline__ bf16_t f2bf(float f) {
  unsigned u = __float_as_uint(f);
  unsigned r = (u + 0x7FFFu + ((u >> 16) & 1u)) >> 16;
  unsigned short h = (unsigned short)r;
  return *reinterpret_cast<bf16_t*>(&h);
}

// ---------------------------------------------------------------------------
// Generic WMMA GEMM:  Y[n, m] = sum_k A[m, k] * B[n, k]   (per batch)
// A: bf16 row-major [M, K] (lda), B: bf16 "n-major" [N, K] (ldb)
// Y: f32 [N, M] (ldy), optional bf16 copy Yb.
// biasMode: 0 none, 1 per-m, 2 per-n.  Optional residual Res[n,m]*mask(n<len).
// Each wave computes a 64x16 tile (4 WMMA accumulators reuse one B fragment).
// Batch index z = blockIdx.y decomposes as (bo = z/Hn, bi = z%Hn).
// ---------------------------------------------------------------------------
__global__ __launch_bounds__(256)
void wmma_gemm_bf16_k(const bf16_t* __restrict__ A, long sA0, long sA1,
                      const bf16_t* __restrict__ Bm, long sB0, long sB1,
                      float* __restrict__ Y, long sY0,
                      bf16_t* __restrict__ Yb, long sYb0, long sYb1,
                      const float* __restrict__ bias, int biasMode,
                      const float* __restrict__ Res, long sR0,
                      const int* __restrict__ lens,
                      int M, int N, int K, int lda, int ldb, int ldy, int Hn)
{
  int lane = threadIdx.x & 31, wv = threadIdx.x >> 5;
  int tilesN = N >> 4;
  int tile = blockIdx.x * 8 + wv;
  if (tile >= (M >> 6) * tilesN) return;          // wave-uniform exit (EXEC stays full)
  int bz = blockIdx.y;
  int bo = bz / Hn, bi = bz - bo * Hn;
  A  += bo * sA0 + bi * sA1;
  Bm += bo * sB0 + bi * sB1;
  if (Y)   Y   += bo * sY0;
  if (Yb)  Yb  += bo * sYb0 + bi * sYb1;
  if (Res) Res += bo * sR0;

  int tm = tile / tilesN;
  int m0 = tm << 6;
  int n0 = (tile - tm * tilesN) << 4;
  int hl = lane >> 4, l15 = lane & 15;

  // B fragment: lane holds column n = n0+l15, K = k0 + 16*hl + e  (32B contiguous)
  const bf16_t* brow = Bm + (long)(n0 + l15) * ldb + (hl << 4);
  // A fragment: lane holds row m, K = k0 + 16*hl + e
  const bf16_t* arow = A + (long)(m0 + l15) * lda + (hl << 4);

  v8f acc[4] = {v8f{}, v8f{}, v8f{}, v8f{}};
  for (int k0 = 0; k0 < K; k0 += 32) {
    v16bf bfrag = *(const v16bf*)(brow + k0);
    const bf16_t* ap = arow + k0;
#pragma unroll
    for (int s = 0; s < 4; ++s) {
      union { v16bf v; v8bf h[2]; } af;
      af.h[0] = *(const v8bf*)(ap);
      af.h[1] = *(const v8bf*)(ap + 8);
      acc[s] = __builtin_amdgcn_wmma_f32_16x16x32_bf16(
          false, af.v, false, bfrag, (short)0, acc[s], false, false);
      ap += (long)16 * lda;
    }
  }

  int nr = n0 + l15;
  float bn = (biasMode == 2) ? bias[nr] : 0.f;
  float rm = 1.f;
  if (lens) rm = (nr < lens[bo]) ? 1.f : 0.f;

#pragma unroll
  for (int s = 0; s < 4; ++s) {
    int mb = m0 + (s << 4) + (hl << 3);          // 8 consecutive m per lane
    float o[8];
#pragma unroll
    for (int r = 0; r < 8; ++r) {
      float v = acc[s][r] + bn;
      if (biasMode == 1) v += bias[mb + r];
      o[r] = v;
    }
    long base = (long)nr * ldy + mb;
    if (Res) {
      const float* rp = Res + base;
#pragma unroll
      for (int r = 0; r < 8; ++r) o[r] += rp[r] * rm;
    }
    if (Y) {
#pragma unroll
      for (int r = 0; r < 8; ++r) Y[base + r] = o[r];
    }
    if (Yb) {
      v8bf pk;
#pragma unroll
      for (int r = 0; r < 8; ++r) pk[r] = f2bf(o[r]);
      *(v8bf*)(Yb + base) = pk;
    }
  }
}

// ---- elementwise / helper kernels -----------------------------------------

__global__ __launch_bounds__(256)
void transpose_f32_k(const float* __restrict__ in, float* __restrict__ out, int T) {
  long i = (long)blockIdx.x * 256 + threadIdx.x;
  long tot = (long)BB * T * CC;
  if (i >= tot) return;
  int c = (int)(i % CC); long r = i / CC; int t = (int)(r % T); int b = (int)(r / T);
  out[i] = in[((long)b * CC + c) * T + t];
}

__global__ __launch_bounds__(256)
void transpose_bf16_k(const float* __restrict__ in, bf16_t* __restrict__ out, int T) {
  long i = (long)blockIdx.x * 256 + threadIdx.x;
  long tot = (long)BB * T * CC;
  if (i >= tot) return;
  int c = (int)(i % CC); long r = i / CC; int t = (int)(r % T); int b = (int)(r / T);
  out[i] = f2bf(in[((long)b * CC + c) * T + t]);
}

__global__ __launch_bounds__(256)
void cvt_bf16_k(const float* __restrict__ in, bf16_t* __restrict__ out, long n) {
  long i = (long)blockIdx.x * 256 + threadIdx.x;
  if (i < n) out[i] = f2bf(in[i]);
}

// conv_w [STK][m][c][3]  ->  Wcat [STK][m][k*CC+c]
__global__ __launch_bounds__(256)
void conv_repack_k(const float* __restrict__ w, bf16_t* __restrict__ out) {
  long i = (long)blockIdx.x * 256 + threadIdx.x;
  long tot = (long)STK * CC * 3 * CC;
  if (i >= tot) return;
  int c = (int)(i % CC); long r = i / CC; int k = (int)(r % 3);
  long r2 = r / 3; int m = (int)(r2 % CC); int s = (int)(r2 / CC);
  out[i] = f2bf(w[(((long)s * CC + m) * CC + c) * 3 + k]);
}

// Xcol[b][t][k*CC+c] = masked x[b][t+k-1][c]
__global__ __launch_bounds__(256)
void im2col_k(const float* __restrict__ X, const int* __restrict__ lens,
              bf16_t* __restrict__ out, int T) {
  long i = (long)blockIdx.x * 256 + threadIdx.x;
  long tot = (long)BB * T * 3 * CC;
  if (i >= tot) return;
  int c = (int)(i % CC); long r = i / CC; int k = (int)(r % 3);
  long r2 = r / 3; int t = (int)(r2 % T); int b = (int)(r2 / T);
  int st = t + k - 1;
  float v = 0.f;
  if (st >= 0 && st < T && st < lens[b]) v = X[((long)b * T + st) * CC + c];
  out[i] = f2bf(v);
}

// per (b,t): LN over C, *g+b, relu, +x ; writes f32 newx and bf16 newx*mask
__global__ __launch_bounds__(256)
void ln_relu_res_k(const float* __restrict__ H, const float* __restrict__ X,
                   const float* __restrict__ g, const float* __restrict__ be,
                   const int* __restrict__ lens,
                   float* __restrict__ XN, bf16_t* __restrict__ XQ, int T) {
  int row = blockIdx.x * 8 + (threadIdx.x >> 5);
  if (row >= BB * T) return;
  int lane = threadIdx.x & 31;
  int b = row / T, t = row - b * T;
  const float* h = H + (long)row * CC;
  float s = 0.f, ss = 0.f;
  for (int c = lane; c < CC; c += 32) { float v = h[c]; s += v; ss += v * v; }
  for (int o = 16; o; o >>= 1) { s += __shfl_xor(s, o, 32); ss += __shfl_xor(ss, o, 32); }
  float mu = s * (1.f / CC);
  float var = ss * (1.f / CC) - mu * mu;
  float rs = rsqrtf(var + 1e-5f);
  float msk = (t < lens[b]) ? 1.f : 0.f;
  const float* x = X + (long)row * CC;
  float* xn = XN + (long)row * CC;
  bf16_t* xq = XQ + (long)row * CC;
  for (int c = lane; c < CC; c += 32) {
    float v = (h[c] - mu) * rs * g[c] + be[c];
    v = fmaxf(v, 0.f) + x[c];
    xn[c] = v;
    xq[c] = f2bf(v * msk);
  }
}

// softmax over d=64 per (b,t,h), * 1/sqrt(64) -> bf16
__global__ __launch_bounds__(256)
void q_softmax_k(const float* __restrict__ Q, bf16_t* __restrict__ O, int T) {
  long i = (long)blockIdx.x * 256 + threadIdx.x;
  long tot = (long)BB * T * NH;
  if (i >= tot) return;
  long bt = i >> 3; int hh = (int)(i & 7);
  const float* q = Q + bt * CC + hh * HD;
  bf16_t* o = O + bt * CC + hh * HD;
  float mx = -3.4e38f;
  for (int d = 0; d < HD; ++d) mx = fmaxf(mx, q[d]);
  float sum = 0.f;
  for (int d = 0; d < HD; ++d) sum += expf(q[d] - mx);
  float inv = 0.125f / sum;
  for (int d = 0; d < HD; ++d) o[d] = f2bf(expf(q[d] - mx) * inv);
}

// softmax over TP=512 per (b,c) row -> bf16
__global__ __launch_bounds__(256)
void k_softmax_k(const float* __restrict__ KC, bf16_t* __restrict__ KS) {
  int row = blockIdx.x * 8 + (threadIdx.x >> 5);
  if (row >= BB * CC) return;
  int lane = threadIdx.x & 31;
  const float* src = KC + (long)row * TPP;
  float v[TPP / 32];
  float mx = -3.4e38f;
#pragma unroll
  for (int j = 0; j < TPP / 32; ++j) { v[j] = src[lane + j * 32]; mx = fmaxf(mx, v[j]); }
  for (int o = 16; o; o >>= 1) mx = fmaxf(mx, __shfl_xor(mx, o, 32));
  float s = 0.f;
#pragma unroll
  for (int j = 0; j < TPP / 32; ++j) { v[j] = expf(v[j] - mx); s += v[j]; }
  for (int o = 16; o; o >>= 1) s += __shfl_xor(s, o, 32);
  float inv = 1.f / s;
  bf16_t* dst = KS + (long)row * TPP;
#pragma unroll
  for (int j = 0; j < TPP / 32; ++j) dst[lane + j * 32] = f2bf(v[j] * inv);
}

// final K=3 conv to 1 channel, mask, optional softplus
__global__ __launch_bounds__(256)
void final_proj_k(const float* __restrict__ X, const int* __restrict__ lens,
                  const float* __restrict__ w, const float* __restrict__ bsc,
                  float* __restrict__ out, int T, int sp) {
  int i = blockIdx.x * 256 + threadIdx.x;
  if (i >= BB * T) return;
  int b = i / T, t = i - b * T;
  int len = lens[b];
  float acc = bsc[0];
  for (int k = 0; k < 3; ++k) {
    int st = t + k - 1;
    if (st < 0 || st >= T || st >= len) continue;
    const float* xr = X + ((long)b * T + st) * CC;
    for (int c = 0; c < CC; ++c) acc += xr[c] * w[c * 3 + k];
  }
  if (t >= len) acc = 0.f;
  if (sp) acc = (acc > 20.f) ? acc : log1pf(expf(acc));
  out[i] = acc;
}

// alignments are one-hot over t for each u: find the index
__global__ __launch_bounds__(256)
void tidx_k(const float* __restrict__ al, int* __restrict__ tidx) {
  int i = blockIdx.x * 256 + threadIdx.x;
  if (i >= BB * TUU) return;
  int b = i / TUU, u = i - b * TUU;
  const float* a = al + (long)b * TTT * TUU + u;
  int t = 0;
  for (int tt = 0; tt < TTT; ++tt) { if (a[(long)tt * TUU] > 0.5f) { t = tt; break; } }
  tidx[i] = t;
}

// length-regulation gather + f0 embedding fused into enc output
__global__ __launch_bounds__(256)
void gather_k(const float* __restrict__ enc, const float* __restrict__ mm,
              const float* __restrict__ ml, const int* __restrict__ tidx,
              const float* __restrict__ f0, const float* __restrict__ few,
              const float* __restrict__ feb,
              float* __restrict__ encT, float* __restrict__ oEnc,
              float* __restrict__ oMm, float* __restrict__ oMl) {
  long i = (long)blockIdx.x * 256 + threadIdx.x;
  long tot = (long)BB * CC * TUU;
  if (i >= tot) return;
  int u = (int)(i % TUU); long r = i / TUU; int c = (int)(r % CC); int b = (int)(r / CC);
  int t = tidx[b * TUU + u];
  long src = ((long)b * CC + c) * TTT + t;
  float e = enc[src];
  encT[((long)b * TUU + u) * CC + c] = e;                 // t-major input for f0 predictor
  oEnc[i] = e + few[c] * f0[(long)b * TUU + u] + feb[c];  // c-major output + embedding
  oMm[i] = mm[src];
  oMl[i] = ml[src];
}

// ---------------------------------------------------------------------------
// Host side
// ---------------------------------------------------------------------------
static inline void gemm(hipStream_t st,
                        const bf16_t* A, long sA0, long sA1,
                        const bf16_t* Bm, long sB0, long sB1,
                        float* Y, long sY0,
                        bf16_t* Yb, long sYb0, long sYb1,
                        const float* bias, int biasMode,
                        const float* Res, long sR0, const int* lens,
                        int M, int N, int K, int lda, int ldb, int ldy,
                        int batches, int Hn) {
  int tiles = (M / 64) * (N / 16);
  dim3 g((unsigned)((tiles + 7) / 8), (unsigned)batches, 1);
  wmma_gemm_bf16_k<<<g, dim3(256, 1, 1), 0, st>>>(A, sA0, sA1, Bm, sB0, sB1, Y, sY0,
                                                  Yb, sYb0, sYb1, bias, biasMode,
                                                  Res, sR0, lens, M, N, K, lda, ldb, ldy, Hn);
}

struct WSBufs {
  bf16_t *wcat, *wq, *wk, *wv, *wo, *Pt, *Xcol, *Xmq, *Qs, *KS, *VB, *CTX, *OUTB;
  float *xA, *Ht, *XN, *KCf;
  int* tidx;
};

static void run_predictor(hipStream_t st, const float* const* P, const int* lens,
                          int T, float* pred_out, int softplus, const WSBufs& w) {
  long nW = (long)STK * CC * CC;
  long nCat = (long)STK * CC * 3 * CC;
  conv_repack_k<<<(unsigned)((nCat + 255) / 256), 256, 0, st>>>(P[0], w.wcat);
  cvt_bf16_k<<<(unsigned)((nW + 255) / 256), 256, 0, st>>>(P[4],  w.wq, nW);
  cvt_bf16_k<<<(unsigned)((nW + 255) / 256), 256, 0, st>>>(P[6],  w.wk, nW);
  cvt_bf16_k<<<(unsigned)((nW + 255) / 256), 256, 0, st>>>(P[8],  w.wv, nW);
  cvt_bf16_k<<<(unsigned)((nW + 255) / 256), 256, 0, st>>>(P[10], w.wo, nW);

  for (int i = 0; i < STK; ++i) {
    long nIm = (long)BB * T * 3 * CC;
    im2col_k<<<(unsigned)((nIm + 255) / 256), 256, 0, st>>>(w.xA, lens, w.Xcol, T);
    // conv GEMM: h = Wcat_i @ Xcol + conv_b[i]
    gemm(st, w.wcat + (long)i * CC * 3 * CC, 0, 0, w.Xcol, (long)T * 3 * CC, 0,
         w.Ht, (long)T * CC, nullptr, 0, 0,
         P[1] + i * CC, 1, nullptr, 0, nullptr,
         CC, T, 3 * CC, 3 * CC, 3 * CC, CC, BB, 1);
    // LN + relu + residual -> XN (f32) and Xmq (bf16 masked)
    ln_relu_res_k<<<(unsigned)(BB * T / 8), 256, 0, st>>>(w.Ht, w.xA, P[2] + i * CC,
                                                          P[3] + i * CC, lens, w.XN, w.Xmq, T);
    // q = qw @ xm + qb  (reuse Ht)
    gemm(st, w.wq + (long)i * CC * CC, 0, 0, w.Xmq, (long)T * CC, 0,
         w.Ht, (long)T * CC, nullptr, 0, 0,
         P[5] + i * CC, 1, nullptr, 0, nullptr,
         CC, T, CC, CC, CC, CC, BB, 1);
    q_softmax_k<<<(unsigned)(((long)BB * T * NH + 255) / 256), 256, 0, st>>>(w.Ht, w.Qs, T);
    // k = kw @ prompts + kb  -> c-major kc (A/B roles swapped)
    gemm(st, w.Pt, (long)TPP * CC, 0, w.wk + (long)i * CC * CC, 0, 0,
         w.KCf, (long)CC * TPP, nullptr, 0, 0,
         P[7] + i * CC, 2, nullptr, 0, nullptr,
         TPP, CC, CC, CC, CC, TPP, BB, 1);
    k_softmax_k<<<(unsigned)(BB * CC / 8), 256, 0, st>>>(w.KCf, w.KS);
    // v = vw @ prompts + vb  -> c-major, bf16 only
    gemm(st, w.Pt, (long)TPP * CC, 0, w.wv + (long)i * CC * CC, 0, 0,
         nullptr, 0, w.VB, (long)CC * TPP, 0,
         P[9] + i * CC, 2, nullptr, 0, nullptr,
         TPP, CC, CC, CC, CC, TPP, BB, 1);
    // ctx[e][d] = sum_n k[d,n] v[e,n]  per (b,h)
    gemm(st, w.KS, (long)CC * TPP, (long)HD * TPP, w.VB, (long)CC * TPP, (long)HD * TPP,
         nullptr, 0, w.CTX, (long)NH * HD * HD, (long)HD * HD,
         nullptr, 0, nullptr, 0, nullptr,
         HD, HD, TPP, TPP, TPP, HD, BB * NH, NH);
    // out[t][e] = sum_d ctx[d,e] q[d,t]  per (b,h), bf16 into OUTB columns h*64..
    gemm(st, w.CTX, (long)NH * HD * HD, (long)HD * HD, w.Qs, (long)T * CC, (long)HD,
         nullptr, 0, w.OUTB, (long)T * CC, (long)HD,
         nullptr, 0, nullptr, 0, nullptr,
         HD, T, HD, HD, CC, CC, BB * NH, NH);
    // x_next = ow @ out + ob + xm   (masked residual)
    gemm(st, w.wo + (long)i * CC * CC, 0, 0, w.OUTB, (long)T * CC, 0,
         w.xA, (long)T * CC, nullptr, 0, 0,
         P[11] + i * CC, 1, w.XN, (long)T * CC, lens,
         CC, T, CC, CC, CC, CC, BB, 1);
  }
  final_proj_k<<<(unsigned)((BB * T + 255) / 256), 256, 0, st>>>(w.xA, lens, P[12], P[13],
                                                                 pred_out, T, softplus);
}

extern "C" void kernel_launch(void* const* d_in, const int* in_sizes, int n_in,
                              void* d_out, int out_size, void* d_ws, size_t ws_size,
                              hipStream_t stream) {
  (void)in_sizes; (void)n_in; (void)out_size; (void)ws_size;
  const float* encodings = (const float*)d_in[0];
  const int*   enc_len   = (const int*)d_in[1];
  const float* prompts   = (const float*)d_in[2];
  const float* mas_means = (const float*)d_in[3];
  const float* mas_lstd  = (const float*)d_in[4];
  const float* align     = (const float*)d_in[5];
  const float* f0s       = (const float*)d_in[6];
  const int*   lat_len   = (const int*)d_in[7];
  const float* durP[14]; for (int i = 0; i < 14; ++i) durP[i] = (const float*)d_in[8 + i];
  const float* f0P[14];  for (int i = 0; i < 14; ++i) f0P[i]  = (const float*)d_in[22 + i];
  const float* few = (const float*)d_in[36];
  const float* feb = (const float*)d_in[37];
  float* fp = (float*)d_out;

  // d_out layout (floats), reference return order
  const long OFF_ENC = 0;
  const long OFF_MM  = OFF_ENC + (long)BB * CC * TUU;
  const long OFF_ML  = OFF_MM  + (long)BB * CC * TUU;
  const long OFF_DUR = OFF_ML  + (long)BB * CC * TUU;
  const long OFF_F0P = OFF_DUR + (long)BB * TTT;
  const long OFF_AL  = OFF_F0P + (long)BB * TUU;
  const long OFF_F0S = OFF_AL  + (long)BB * TTT * TUU;
  const long OFF_LL  = OFF_F0S + (long)BB * TUU;

  // bump allocator over d_ws
  size_t off = 0;
  auto alloc = [&](size_t bytes) -> void* {
    off = (off + 255) & ~(size_t)255;
    void* p = (char*)d_ws + off;
    off += bytes;
    return p;
  };
  WSBufs w;
  const int Tmax = TUU;
  w.wcat = (bf16_t*)alloc((size_t)STK * CC * 3 * CC * 2);
  w.wq   = (bf16_t*)alloc((size_t)STK * CC * CC * 2);
  w.wk   = (bf16_t*)alloc((size_t)STK * CC * CC * 2);
  w.wv   = (bf16_t*)alloc((size_t)STK * CC * CC * 2);
  w.wo   = (bf16_t*)alloc((size_t)STK * CC * CC * 2);
  w.Pt   = (bf16_t*)alloc((size_t)BB * TPP * CC * 2);
  w.Xcol = (bf16_t*)alloc((size_t)BB * Tmax * 3 * CC * 2);
  w.Xmq  = (bf16_t*)alloc((size_t)BB * Tmax * CC * 2);
  w.Qs   = (bf16_t*)alloc((size_t)BB * Tmax * CC * 2);
  w.KS   = (bf16_t*)alloc((size_t)BB * CC * TPP * 2);
  w.VB   = (bf16_t*)alloc((size_t)BB * CC * TPP * 2);
  w.CTX  = (bf16_t*)alloc((size_t)BB * NH * HD * HD * 2);
  w.OUTB = (bf16_t*)alloc((size_t)BB * Tmax * CC * 2);
  w.xA   = (float*)alloc((size_t)BB * Tmax * CC * 4);
  w.Ht   = (float*)alloc((size_t)BB * Tmax * CC * 4);
  w.XN   = (float*)alloc((size_t)BB * Tmax * CC * 4);
  w.KCf  = (float*)alloc((size_t)BB * CC * TPP * 4);
  w.tidx = (int*)alloc((size_t)BB * TUU * 4);

  // prompts -> t-major bf16 (shared by both predictors)
  {
    long n = (long)BB * TPP * CC;
    transpose_bf16_k<<<(unsigned)((n + 255) / 256), 256, 0, stream>>>(prompts, w.Pt, TPP);
  }
  // duration predictor on encodings (T = 256)
  {
    long n = (long)BB * TTT * CC;
    transpose_f32_k<<<(unsigned)((n + 255) / 256), 256, 0, stream>>>(encodings, w.xA, TTT);
  }
  run_predictor(stream, durP, enc_len, TTT, fp + OFF_DUR, 1, w);

  // length regulation: one-hot gather + f0 embedding; fills xA (t-major) for f0 predictor
  tidx_k<<<(unsigned)((BB * TUU + 255) / 256), 256, 0, stream>>>(align, w.tidx);
  {
    long n = (long)BB * CC * TUU;
    gather_k<<<(unsigned)((n + 255) / 256), 256, 0, stream>>>(
        encodings, mas_means, mas_lstd, w.tidx, f0s, few, feb,
        w.xA, fp + OFF_ENC, fp + OFF_MM, fp + OFF_ML);
  }
  // f0 predictor on regulated encodings (T = 2048)
  run_predictor(stream, f0P, lat_len, TUU, fp + OFF_F0P, 0, w);

  // pass-through outputs (raw copies preserve exact bits)
  hipMemcpyAsync(fp + OFF_AL,  align,   sizeof(float) * (size_t)BB * TTT * TUU,
                 hipMemcpyDeviceToDevice, stream);
  hipMemcpyAsync(fp + OFF_F0S, f0s,     sizeof(float) * (size_t)BB * TUU,
                 hipMemcpyDeviceToDevice, stream);
  hipMemcpyAsync(fp + OFF_LL,  lat_len, sizeof(int) * BB,
                 hipMemcpyDeviceToDevice, stream);
}
